// SelfAttention_5609227288809
// MI455X (gfx1250) — matrix operand order repaired and checked by reading, not compiled, not run on hardware
//
#include <hip/hip_runtime.h>
#include <hip/hip_bf16.h>

#define DEV __device__ __forceinline__

typedef _Float16 half_t;
typedef __attribute__((ext_vector_type(4)))  _Float16 v4h;
typedef __attribute__((ext_vector_type(8)))  _Float16 v8h;
typedef __attribute__((ext_vector_type(16))) _Float16 v16h;
typedef __attribute__((ext_vector_type(4)))  float    v4f;
typedef __attribute__((ext_vector_type(8)))  float    v8f;

static constexpr int kB  = 16;
static constexpr int kS  = 2048;
static constexpr int kD  = 1024;
static constexpr int kBS = kB * kS;   // 32768 rows

// ---------------------------------------------------------------------------
// WMMA fragment loaders (wave32; A and B share one contiguous K order chosen here)
// A (16x32 f16): lane m=lane&15 holds row m; halves 0..7 = K kb..kb+7,
//                halves 8..15 = K kb+8..kb+15, kb = 16*(lane>>4).
// B (32x16 f16): operand stored as [N rows x K cols] row-major; lane n=lane&15
//                holds "column" n; group g=lane>>4 holds K = k0+16g .. +15
//                => one contiguous 32-byte load.
// C/D (16x16 f32): vgpr r, lane: row = r + 8*(lane>>4), col = lane&15.
// ---------------------------------------------------------------------------
DEV v16h load_frag_a(const half_t* __restrict__ base, int ld, int k0, int lane) {
  const half_t* p = base + (size_t)(lane & 15) * ld + k0 + ((lane >> 4) << 4);
  union { v16h v; v8h h[2]; } u;
  u.h[0] = *(const v8h*)(p);        // K = kb .. kb+7
  u.h[1] = *(const v8h*)(p + 8);    // K = kb+8 .. kb+15
  return u.v;
}

DEV v16h load_frag_bt(const half_t* __restrict__ base, int ld, int k0, int lane) {
  const half_t* p = base + (size_t)(lane & 15) * ld + k0 + ((lane >> 4) << 4);
  return *(const v16h*)(p);         // K = k0+16g .. k0+16g+15 (32B contiguous)
}

DEV v8f wmma_f16(v16h a, v16h b, v8f c) {
  return __builtin_amdgcn_wmma_f32_16x16x32_f16(false, a, false, b, (short)0, c,
                                                false, false);
}

DEV float rmax16(float v) {
  #pragma unroll
  for (int off = 1; off < 16; off <<= 1) v = fmaxf(v, __shfl_xor(v, off, 32));
  return v;
}
DEV float rsum16(float v) {
  #pragma unroll
  for (int off = 1; off < 16; off <<= 1) v += __shfl_xor(v, off, 32);
  return v;
}

// ---------------------------------------------------------------------------
// 0) zero p-bar
// ---------------------------------------------------------------------------
__global__ void zero_f32(float* __restrict__ p, int n) {
  int i = blockIdx.x * blockDim.x + threadIdx.x;
  if (i < n) p[i] = 0.0f;
}

// ---------------------------------------------------------------------------
// 1) embedding gather (padding_idx=0) -> f16 H [BS, D]
// ---------------------------------------------------------------------------
__global__ void gather_embed(const int* __restrict__ X, const float* __restrict__ emb,
                             half_t* __restrict__ H) {
  int row = blockIdx.x;                 // 0..BS-1
  int tok = X[row];
  half_t* dst = H + (size_t)row * kD;
  int i = threadIdx.x * 4;              // 256 threads x 4 = 1024
  if (tok == 0) {
    v4h z = {};
    *(v4h*)(dst + i) = z;
  } else {
    v4f f = *(const v4f*)(emb + (size_t)tok * kD + i);
    v4h h;
    h[0] = (half_t)f[0]; h[1] = (half_t)f[1]; h[2] = (half_t)f[2]; h[3] = (half_t)f[3];
    *(v4h*)(dst + i) = h;
  }
}

// ---------------------------------------------------------------------------
// 2) weight transpose+convert: w [K,N] f32 -> wT [N,K] f16 (B-operand form)
// ---------------------------------------------------------------------------
__global__ void transpose_w(const float* __restrict__ w, half_t* __restrict__ wT) {
  __shared__ float tile[16][17];
  int kb = blockIdx.x * 16, nb = blockIdx.y * 16;
  int tx = threadIdx.x, ty = threadIdx.y;
  tile[ty][tx] = w[(size_t)(kb + ty) * kD + nb + tx];
  __syncthreads();
  wT[(size_t)(nb + ty) * kD + kb + tx] = (half_t)tile[tx][ty];
}

// ---------------------------------------------------------------------------
// 3) GEMM: out[M,D] = (A[M,D] @ W + bias) * scale, f16 out.
//    Register-blocked: each wave computes a 32x64 tile (2 A-frags x 4 B-frags,
//    8 WMMA per K-step; A reused 4x, B reused 2x => 21 FLOP/byte from L2).
// ---------------------------------------------------------------------------
__global__ void __launch_bounds__(128)
gemm_wmma(const half_t* __restrict__ A, const half_t* __restrict__ Wt,
          const float* __restrict__ bias, half_t* __restrict__ out, float scale) {
  int lane = threadIdx.x & 31;
  int wave = threadIdx.x >> 5;
  int mt = blockIdx.x;                  // 32-row tile, 0..kBS/32-1
  int ng = blockIdx.y * 4 + wave;       // 64-col group, 0..kD/64-1
  const half_t* A0 = A + (size_t)mt * 32 * kD;
  const half_t* A1 = A0 + (size_t)16 * kD;
  const half_t* Bg = Wt + (size_t)ng * 64 * kD;

  v8f c[2][4];
  #pragma unroll
  for (int i = 0; i < 2; ++i)
    #pragma unroll
    for (int j = 0; j < 4; ++j) c[i][j] = (v8f){};

  for (int k0 = 0; k0 < kD; k0 += 32) {
    v16h a0 = load_frag_a(A0, kD, k0, lane);
    v16h a1 = load_frag_a(A1, kD, k0, lane);
    #pragma unroll
    for (int j = 0; j < 4; ++j) {
      v16h b = load_frag_bt(Bg + (size_t)j * 16 * kD, kD, k0, lane);
      c[0][j] = wmma_f16(a0, b, c[0][j]);
      c[1][j] = wmma_f16(a1, b, c[1][j]);
    }
  }

  #pragma unroll
  for (int i = 0; i < 2; ++i) {
    int rbase = mt * 32 + i * 16 + ((lane >> 4) << 3);
    #pragma unroll
    for (int j = 0; j < 4; ++j) {
      int n = ng * 64 + j * 16 + (lane & 15);
      float bvn = bias[n];
      #pragma unroll
      for (int r = 0; r < 8; ++r) {
        float v = (c[i][j][r] + bvn) * scale;
        out[(size_t)(rbase + r) * kD + n] = (half_t)v;
      }
    }
  }
}

// ---------------------------------------------------------------------------
// 4) attention: per 16-query tile (one wave), two passes over all key tiles,
//    4 key tiles per sweep (Q fragment reused 4x per load):
//    pass 1: online row max M and sum-exp L;  pass 2: accumulate column means
//    p̄[b,k] += (1/S) * sum_q exp(s-M)/L   via atomics.
// ---------------------------------------------------------------------------
__global__ void __launch_bounds__(128)
attn_pbar(const half_t* __restrict__ Q, const half_t* __restrict__ Kmat,
          float* __restrict__ pbar) {
  int lane = threadIdx.x & 31;
  int wave = threadIdx.x >> 5;
  int qt = blockIdx.x * 4 + wave;       // 0 .. B*S/16-1
  int b  = qt >> 7;                     // S/16 = 128 tiles per batch
  int q0 = (qt & 127) << 4;
  const half_t* Qb = Q    + (size_t)(b * kS + q0) * kD;
  const half_t* Kb = Kmat + (size_t)b * kS * kD;

  float Mr[8], Lr[8];
  #pragma unroll
  for (int r = 0; r < 8; ++r) { Mr[r] = -1e30f; Lr[r] = 0.0f; }

  // ---- pass 1: stats (Q already carries the 1/sqrt(D) scale + bias) ----
  for (int kt = 0; kt < kS; kt += 64) {
    v8f c[4];
    #pragma unroll
    for (int j = 0; j < 4; ++j) c[j] = (v8f){};
    for (int d0 = 0; d0 < kD; d0 += 32) {
      v16h a = load_frag_a(Qb, kD, d0, lane);
      #pragma unroll
      for (int j = 0; j < 4; ++j) {
        v16h bt = load_frag_bt(Kb + (size_t)(kt + j * 16) * kD, kD, d0, lane);
        c[j] = wmma_f16(a, bt, c[j]);
      }
    }
    #pragma unroll
    for (int r = 0; r < 8; ++r) {
      float tmax = fmaxf(fmaxf(c[0][r], c[1][r]), fmaxf(c[2][r], c[3][r]));
      tmax = rmax16(tmax);
      float mnew = fmaxf(Mr[r], tmax);
      float ps = __expf(c[0][r] - mnew) + __expf(c[1][r] - mnew) +
                 __expf(c[2][r] - mnew) + __expf(c[3][r] - mnew);
      ps = rsum16(ps);
      Lr[r] = Lr[r] * __expf(Mr[r] - mnew) + ps;
      Mr[r] = mnew;
    }
  }
  float inv[8];
  #pragma unroll
  for (int r = 0; r < 8; ++r) inv[r] = 1.0f / Lr[r];

  // ---- pass 2: accumulate column means of P into p̄ ----
  const float wmean = 1.0f / (float)kS;
  for (int kt = 0; kt < kS; kt += 64) {
    v8f c[4];
    #pragma unroll
    for (int j = 0; j < 4; ++j) c[j] = (v8f){};
    for (int d0 = 0; d0 < kD; d0 += 32) {
      v16h a = load_frag_a(Qb, kD, d0, lane);
      #pragma unroll
      for (int j = 0; j < 4; ++j) {
        v16h bt = load_frag_bt(Kb + (size_t)(kt + j * 16) * kD, kD, d0, lane);
        c[j] = wmma_f16(a, bt, c[j]);
      }
    }
    #pragma unroll
    for (int j = 0; j < 4; ++j) {
      float colp = 0.0f;
      #pragma unroll
      for (int r = 0; r < 8; ++r) colp += __expf(c[j][r] - Mr[r]) * inv[r];
      colp += __shfl_xor(colp, 16, 32);  // rows 0-7 (+) rows 8-15 of this column
      if (lane < 16)
        atomicAdd(&pbar[b * kS + kt + j * 16 + lane], colp * wmean);
    }
  }
}

// ---------------------------------------------------------------------------
// 5) t[b,:] = p̄[b,:] @ H_b      ([1xS] x [SxD])
// ---------------------------------------------------------------------------
__global__ void __launch_bounds__(256)
gemv_pbar_h(const float* __restrict__ pbar, const half_t* __restrict__ H,
            float* __restrict__ t) {
  int b = blockIdx.x, tid = threadIdx.x;
  const float* pb  = pbar + (size_t)b * kS;
  const half_t* Hb = H    + (size_t)b * kS * kD;
  float a0 = 0, a1 = 0, a2 = 0, a3 = 0;
  for (int s = 0; s < kS; ++s) {
    float p = pb[s];
    const half_t* hr = Hb + (size_t)s * kD;
    a0 += p * (float)hr[tid];
    a1 += p * (float)hr[tid + 256];
    a2 += p * (float)hr[tid + 512];
    a3 += p * (float)hr[tid + 768];
  }
  float* tb = t + (size_t)b * kD;
  tb[tid] = a0; tb[tid + 256] = a1; tb[tid + 512] = a2; tb[tid + 768] = a3;
}

// ---------------------------------------------------------------------------
// 6) out[b,:] = t[b,:] @ wv + bv   (sum_k p̄[k] == 1 so +bv is exact)
// ---------------------------------------------------------------------------
__global__ void __launch_bounds__(256)
gemv_t_wv(const float* __restrict__ t, const float* __restrict__ wv,
          const float* __restrict__ bv, float* __restrict__ out) {
  int b = blockIdx.x, tid = threadIdx.x;
  float a0 = bv[tid], a1 = bv[tid + 256], a2 = bv[tid + 512], a3 = bv[tid + 768];
  const float* tb = t + (size_t)b * kD;
  for (int k = 0; k < kD; ++k) {
    float tv = tb[k];
    const float* wr = wv + (size_t)k * kD;
    a0 += tv * wr[tid];
    a1 += tv * wr[tid + 256];
    a2 += tv * wr[tid + 512];
    a3 += tv * wr[tid + 768];
  }
  float* ob = out + (size_t)b * kD;
  ob[tid] = a0; ob[tid + 256] = a1; ob[tid + 512] = a2; ob[tid + 768] = a3;
}

// ---------------------------------------------------------------------------
extern "C" void kernel_launch(void* const* d_in, const int* in_sizes, int n_in,
                              void* d_out, int out_size, void* d_ws, size_t ws_size,
                              hipStream_t stream) {
  (void)in_sizes; (void)n_in; (void)out_size; (void)ws_size;
  const int*   X   = (const int*)  d_in[0];
  const float* emb = (const float*)d_in[1];
  const float* wq  = (const float*)d_in[2];
  const float* bq  = (const float*)d_in[3];
  const float* wk  = (const float*)d_in[4];
  const float* bk  = (const float*)d_in[5];
  const float* wv  = (const float*)d_in[6];
  const float* bv  = (const float*)d_in[7];
  float* out = (float*)d_out;

  uint8_t* base = (uint8_t*)d_ws;
  size_t off = 0;
  auto take = [&](size_t bytes) -> void* {
    void* p = base + off;
    off = (off + bytes + 255) & ~(size_t)255;
    return p;
  };
  half_t* H    = (half_t*)take((size_t)kBS * kD * 2);   // 64 MB
  half_t* Qm   = (half_t*)take((size_t)kBS * kD * 2);   // 64 MB
  half_t* Km   = (half_t*)take((size_t)kBS * kD * 2);   // 64 MB
  half_t* wqT  = (half_t*)take((size_t)kD * kD * 2);    //  2 MB
  half_t* wkT  = (half_t*)take((size_t)kD * kD * 2);    //  2 MB
  float*  pbar = (float*) take((size_t)kBS * 4);        // 128 KB
  float*  t    = (float*) take((size_t)kB * kD * 4);    //  64 KB

  zero_f32<<<kBS / 256, 256, 0, stream>>>(pbar, kBS);
  gather_embed<<<kBS, 256, 0, stream>>>(X, emb, H);
  transpose_w<<<dim3(kD / 16, kD / 16), dim3(16, 16), 0, stream>>>(wq, wqT);
  transpose_w<<<dim3(kD / 16, kD / 16), dim3(16, 16), 0, stream>>>(wk, wkT);

  // Q carries bias and the 1/sqrt(D)=1/32 score scale; K carries bias only.
  gemm_wmma<<<dim3(kBS / 32, kD / 256), 128, 0, stream>>>(H, wqT, bq, Qm, 1.0f / 32.0f);
  gemm_wmma<<<dim3(kBS / 32, kD / 256), 128, 0, stream>>>(H, wkT, bk, Km, 1.0f);

  attn_pbar<<<(kB * kS / 16) / 4, 128, 0, stream>>>(Qm, Km, pbar);

  gemv_pbar_h<<<kB, 256, 0, stream>>>(pbar, H, t);
  gemv_t_wv<<<kB, 256, 0, stream>>>(t, wv, bv, out);
}
